// NONLocal_block_37014028157091
// MI455X (gfx1250) — matrix operand order repaired and checked by reading, not compiled, not run on hardware
//
#include <hip/hip_runtime.h>
#include <stdint.h>

#define BB 4
#define CC 64
#define LL 4096

typedef __attribute__((ext_vector_type(16))) __bf16 v16bf;
typedef __attribute__((ext_vector_type(8)))  float  v8f;

union BF16x16 { v16bf v; uint4 q[2]; };

__device__ __forceinline__ __bf16 f2bf(float f) {
    union { float f; uint32_t u; } cin; cin.f = f;
    uint32_t r = (cin.u + 0x7FFFu + ((cin.u >> 16) & 1u)) >> 16;  // RNE
    union { uint16_t u; __bf16 h; } cout; cout.u = (uint16_t)r;
    return cout.h;
}

// 16-lane xor-shuffle on the VALU via v_permlane16_b32 (no LDS-pipe traffic).
// sel constants: nibble i = (i ^ mask) for lanes 0..7 (s1) and 8..15 (s2), row-relative.
__device__ __forceinline__ float permxor_f(float v, unsigned s1, unsigned s2) {
    unsigned u = __builtin_bit_cast(unsigned, v);
    unsigned p = __builtin_amdgcn_permlane16(u, u, s1, s2, false, false);
    return __builtin_bit_cast(float, p);
}
__device__ __forceinline__ float rowmax16(float v) {
    v = fmaxf(v, permxor_f(v, 0x67452301u, 0xEFCDAB89u)); // xor 1
    v = fmaxf(v, permxor_f(v, 0x54761032u, 0xDCFE98BAu)); // xor 2
    v = fmaxf(v, permxor_f(v, 0x32107654u, 0xBA98FEDCu)); // xor 4
    v = fmaxf(v, permxor_f(v, 0xFEDCBA98u, 0x76543210u)); // xor 8
    return v;
}
__device__ __forceinline__ float rowsum16(float v) {
    v += permxor_f(v, 0x67452301u, 0xEFCDAB89u);
    v += permxor_f(v, 0x54761032u, 0xDCFE98BAu);
    v += permxor_f(v, 0x32107654u, 0xBA98FEDCu);
    v += permxor_f(v, 0xFEDCBA98u, 0x76543210u);
    return v;
}

// ---------------- zero BN stats ----------------
__global__ void zero_stats_kernel(float* stats) {
    if (threadIdx.x < 2 * CC) stats[threadIdx.x] = 0.f;
}

// ------------- 1x1 conv projection, fp32 in -> bf16 out -------------
// mode 0: out[(b*LL+pos)*64 + c]   (position-major rows, channel contiguous)
// mode 1: out[(b*CC+c)*LL + pos]   (channel-major rows, position contiguous)
__global__ __launch_bounds__(256)
void proj_kernel(const float* __restrict__ inp, const float* __restrict__ w,
                 const float* __restrict__ bias, __bf16* __restrict__ out, int mode)
{
    int b  = blockIdx.x >> 6;
    int p0 = (blockIdx.x & 63) << 6;
    __shared__ float xs[CC][65];
    int tid = threadIdx.x;
    for (int idx = tid; idx < CC * 64; idx += 256) {
        int ci = idx >> 6, j = idx & 63;
        xs[ci][j] = inp[(size_t)(b * CC + ci) * LL + p0 + j];
    }
    __syncthreads();
    int c = tid & 63, jb = tid >> 6;            // c = out channel, jb = position sub-slot
    float acc[16];
    float bv = bias[c];
    #pragma unroll
    for (int j = 0; j < 16; ++j) acc[j] = bv;
    for (int k = 0; k < CC; ++k) {
        float wk = w[c * CC + k];
        #pragma unroll
        for (int j = 0; j < 16; ++j) acc[j] += wk * xs[k][jb + 4 * j];
    }
    #pragma unroll
    for (int j = 0; j < 16; ++j) {
        int pos = p0 + jb + 4 * j;
        __bf16 v = f2bf(acc[j]);
        if (mode == 0) out[((size_t)(b * LL + pos) << 6) + c] = v;
        else           out[(size_t)(b * CC + c) * LL + pos]   = v;
    }
}

// ------------- fused flash-attention core: S = theta*phi^T, online softmax, y += P*g -------------
// thetaT: [b][n][c] bf16, phiT: [b][m][c] bf16, gT: [b][c][m] bf16, yout: [b][n][c] f32
__global__ __launch_bounds__(128)
void flash_kernel(const __bf16* __restrict__ thetaT, const __bf16* __restrict__ phiT,
                  const __bf16* __restrict__ gT, float* __restrict__ yout)
{
    __shared__ __align__(16) __bf16 pst[4][16][32];   // per-wave P staging tile
    int b    = blockIdx.x >> 6;
    int n0   = (blockIdx.x & 63) << 6;
    int wv   = threadIdx.x >> 5;
    int lane = threadIdx.x & 31;
    int half = lane >> 4;       // 0: K 0..7,16..23 side, 1: K 8..15,24..31 side
    int coln = lane & 15;
    int klo  = half ? 8 : 0;    // A-fragment K sub-offset

    // theta A-fragments (16 rows x 64 K), two K-chunks of 32
    BF16x16 a0, a1;
    {
        const __bf16* trow = thetaT + (size_t)(b * LL + n0 + wv * 16 + coln) * CC;
        a0.q[0] = *(const uint4*)(trow + klo);
        a0.q[1] = *(const uint4*)(trow + 16 + klo);
        a1.q[0] = *(const uint4*)(trow + 32 + klo);
        a1.q[1] = *(const uint4*)(trow + 48 + klo);
    }

    v8f zero = {};
    v8f yac0 = zero, yac1 = zero, yac2 = zero, yac3 = zero;  // 16 n x 64 c accumulators
    float mrun[8], lrun[8];
    #pragma unroll
    for (int r = 0; r < 8; ++r) { mrun[r] = -1e30f; lrun[r] = 0.f; }

    const __bf16* phiBase0 = phiT + (size_t)(b * LL + coln)      * CC + (half << 3);
    const __bf16* phiBase1 = phiT + (size_t)(b * LL + 16 + coln) * CC + (half << 3);
    const __bf16* gBase    = gT + (size_t)b * CC * LL + (half << 3);

    for (int m0 = 0; m0 < LL; m0 += 32) {
        // phi B-fragments: 2 m-column groups x 2 K-chunks
        const __bf16* prow0 = phiBase0 + (size_t)m0 * CC;
        const __bf16* prow1 = phiBase1 + (size_t)m0 * CC;
        BF16x16 bp0, bp1, bp2, bp3;
        bp0.q[0] = *(const uint4*)(prow0);      bp0.q[1] = *(const uint4*)(prow0 + 16);
        bp1.q[0] = *(const uint4*)(prow0 + 32); bp1.q[1] = *(const uint4*)(prow0 + 48);
        bp2.q[0] = *(const uint4*)(prow1);      bp2.q[1] = *(const uint4*)(prow1 + 16);
        bp3.q[0] = *(const uint4*)(prow1 + 32); bp3.q[1] = *(const uint4*)(prow1 + 48);

        if (m0 + 32 < LL) {
            __builtin_prefetch((const void*)(prow0 + 32 * CC), 0, 1);
            __builtin_prefetch((const void*)(prow1 + 32 * CC), 0, 1);
        }

        v8f s0 = zero, s1 = zero;
        s0 = __builtin_amdgcn_wmma_f32_16x16x32_bf16(false, a0.v, false, bp0.v, (short)0, s0, false, false);
        s0 = __builtin_amdgcn_wmma_f32_16x16x32_bf16(false, a1.v, false, bp1.v, (short)0, s0, false, false);
        s1 = __builtin_amdgcn_wmma_f32_16x16x32_bf16(false, a0.v, false, bp2.v, (short)0, s1, false, false);
        s1 = __builtin_amdgcn_wmma_f32_16x16x32_bf16(false, a1.v, false, bp3.v, (short)0, s1, false, false);

        // g B-fragments: K = 32 m, 4 groups of 16 channels
        BF16x16 gb0, gb1, gb2, gb3;
        {
            const __bf16* g0 = gBase + (size_t)( 0 + coln) * LL + m0;
            const __bf16* g1 = gBase + (size_t)(16 + coln) * LL + m0;
            const __bf16* g2 = gBase + (size_t)(32 + coln) * LL + m0;
            const __bf16* g3 = gBase + (size_t)(48 + coln) * LL + m0;
            gb0.q[0] = *(const uint4*)g0; gb0.q[1] = *(const uint4*)(g0 + 16);
            gb1.q[0] = *(const uint4*)g1; gb1.q[1] = *(const uint4*)(g1 + 16);
            gb2.q[0] = *(const uint4*)g2; gb2.q[1] = *(const uint4*)(g2 + 16);
            gb3.q[0] = *(const uint4*)g3; gb3.q[1] = *(const uint4*)(g3 + 16);
        }

        // Lazy online softmax. mrun[r] is always row-uniform (result of a 16-lane
        // reduce), so when no lane in the row exceeds it, s - mrun <= 0 and the
        // expensive reduce+rescale can be skipped (wave-uniform branch).
        float p0v[8], p1v[8];
        #pragma unroll
        for (int r = 0; r < 8; ++r) {
            float mxl = fmaxf(s0[r], s1[r]);
            if (__any(mxl > mrun[r])) {
                float mnew  = fmaxf(mrun[r], rowmax16(mxl));
                float alpha = __expf(mrun[r] - mnew);
                lrun[r] *= alpha;
                yac0[r] *= alpha; yac1[r] *= alpha; yac2[r] *= alpha; yac3[r] *= alpha;
                mrun[r] = mnew;
            }
            float e0 = __expf(s0[r] - mrun[r]);
            float e1 = __expf(s1[r] - mrun[r]);
            lrun[r] += e0 + e1;   // per-lane partial, reduced at end
            p0v[r] = e0; p1v[r] = e1;
        }

        // C-fragment -> A-fragment re-layout for P via wave-private LDS tile
        #pragma unroll
        for (int r = 0; r < 8; ++r) {
            pst[wv][r + (half << 3)][coln]      = f2bf(p0v[r]);
            pst[wv][r + (half << 3)][16 + coln] = f2bf(p1v[r]);
        }
        BF16x16 pa;
        {
            const __bf16* pr = &pst[wv][coln][klo];
            pa.q[0] = *(const uint4*)(pr);
            pa.q[1] = *(const uint4*)(pr + 16);
        }
        yac0 = __builtin_amdgcn_wmma_f32_16x16x32_bf16(false, pa.v, false, gb0.v, (short)0, yac0, false, false);
        yac1 = __builtin_amdgcn_wmma_f32_16x16x32_bf16(false, pa.v, false, gb1.v, (short)0, yac1, false, false);
        yac2 = __builtin_amdgcn_wmma_f32_16x16x32_bf16(false, pa.v, false, gb2.v, (short)0, yac2, false, false);
        yac3 = __builtin_amdgcn_wmma_f32_16x16x32_bf16(false, pa.v, false, gb3.v, (short)0, yac3, false, false);
    }

    #pragma unroll
    for (int r = 0; r < 8; ++r) lrun[r] = rowsum16(lrun[r]);
    #pragma unroll
    for (int r = 0; r < 8; ++r) {
        int n = n0 + wv * 16 + r + (half << 3);
        float inv = 1.f / lrun[r];
        float* yr = yout + ((size_t)(b * LL + n) << 6);
        yr[ 0 + coln] = yac0[r] * inv;
        yr[16 + coln] = yac1[r] * inv;
        yr[32 + coln] = yac2[r] * inv;
        yr[48 + coln] = yac3[r] * inv;
    }
}

// ------------- W projection + BN statistics -------------
// y: [b][n][c] f32 -> wy: [b][n][co] f32 (pre-norm), stats[0..63]=sum, [64..127]=sumsq
__global__ __launch_bounds__(256)
void wconv_kernel(const float* __restrict__ y, const float* __restrict__ Ww,
                  const float* __restrict__ Wb, float* __restrict__ wy,
                  float* __restrict__ stats)
{
    int b  = blockIdx.x >> 6;
    int n0 = (blockIdx.x & 63) << 6;
    __shared__ float ys[64][65];
    __shared__ float ssum[4][64];
    __shared__ float ssq[4][64];
    int tid = threadIdx.x;
    for (int idx = tid; idx < 64 * 64; idx += 256) {
        int j = idx >> 6, c = idx & 63;
        ys[j][c] = y[((size_t)(b * LL + n0 + j) << 6) + c];
    }
    __syncthreads();
    int co = tid & 63, jb = tid >> 6;
    float acc[16];
    float bv = Wb[co];
    #pragma unroll
    for (int j = 0; j < 16; ++j) acc[j] = bv;
    for (int k = 0; k < 64; ++k) {
        float wk = Ww[co * 64 + k];
        #pragma unroll
        for (int j = 0; j < 16; ++j) acc[j] += wk * ys[jb + 4 * j][k];
    }
    float s = 0.f, q = 0.f;
    #pragma unroll
    for (int j = 0; j < 16; ++j) {
        int n = n0 + jb + 4 * j;
        wy[((size_t)(b * LL + n) << 6) + co] = acc[j];
        s += acc[j]; q += acc[j] * acc[j];
    }
    ssum[jb][co] = s; ssq[jb][co] = q;
    __syncthreads();
    if (jb == 0) {
        float ts = ssum[0][co] + ssum[1][co] + ssum[2][co] + ssum[3][co];
        float tq = ssq[0][co] + ssq[1][co] + ssq[2][co] + ssq[3][co];
        atomicAdd(&stats[co], ts);
        atomicAdd(&stats[64 + co], tq);
    }
}

// ------------- batchnorm (training stats) + residual -------------
__global__ __launch_bounds__(256)
void finalize_kernel(const float* __restrict__ wy, const float* __restrict__ stats,
                     const float* __restrict__ gamma, const float* __restrict__ beta,
                     const float* __restrict__ x, float* __restrict__ out)
{
    int idx = blockIdx.x * 256 + threadIdx.x;      // idx = (b*64 + co)*4096 + n
    int n  = idx & (LL - 1);
    int co = (idx >> 12) & 63;
    int b  = idx >> 18;
    const float invBN = 1.f / (float)(BB * LL);
    float mean = stats[co] * invBN;
    float var  = stats[64 + co] * invBN - mean * mean;
    float v = wy[((size_t)(b * LL + n) << 6) + co];
    out[idx] = gamma[co] * (v - mean) * rsqrtf(var + 1e-5f) + beta[co] + x[idx];
}

extern "C" void kernel_launch(void* const* d_in, const int* in_sizes, int n_in,
                              void* d_out, int out_size, void* d_ws, size_t ws_size,
                              hipStream_t stream)
{
    (void)in_sizes; (void)n_in; (void)out_size; (void)ws_size;
    const float* x    = (const float*)d_in[0];
    const float* x2   = (const float*)d_in[1];
    const float* g_w  = (const float*)d_in[2];
    const float* g_b  = (const float*)d_in[3];
    const float* th_w = (const float*)d_in[4];
    const float* th_b = (const float*)d_in[5];
    const float* ph_w = (const float*)d_in[6];
    const float* ph_b = (const float*)d_in[7];
    const float* W_w  = (const float*)d_in[8];
    const float* W_b  = (const float*)d_in[9];
    const float* gam  = (const float*)d_in[10];
    const float* bet  = (const float*)d_in[11];
    float* out = (float*)d_out;

    // workspace layout (14 MB + 512 B total)
    char* ws = (char*)d_ws;
    __bf16* thetaT = (__bf16*)(ws);                               // 2 MB  [b][n][c]
    __bf16* phiT   = (__bf16*)(ws + (size_t)2  * 1024 * 1024);    // 2 MB  [b][m][c]
    __bf16* gT     = (__bf16*)(ws + (size_t)4  * 1024 * 1024);    // 2 MB  [b][c][m]
    float*  yws    = (float*) (ws + (size_t)6  * 1024 * 1024);    // 4 MB  [b][n][c]
    float*  wyws   = (float*) (ws + (size_t)10 * 1024 * 1024);    // 4 MB  [b][n][co]
    float*  stats  = (float*) (ws + (size_t)14 * 1024 * 1024);    // 512 B

    zero_stats_kernel<<<1, 128, 0, stream>>>(stats);
    proj_kernel<<<BB * (LL / 64), 256, 0, stream>>>(x,  th_w, th_b, thetaT, 0);
    proj_kernel<<<BB * (LL / 64), 256, 0, stream>>>(x2, ph_w, ph_b, phiT, 0);
    proj_kernel<<<BB * (LL / 64), 256, 0, stream>>>(x2, g_w,  g_b,  gT, 1);
    flash_kernel<<<BB * (LL / 64), 128, 0, stream>>>(thetaT, phiT, gT, yws);
    wconv_kernel<<<BB * (LL / 64), 256, 0, stream>>>(yws, W_w, W_b, wyws, stats);
    finalize_kernel<<<(BB * CC * LL) / 256, 256, 0, stream>>>(wyws, stats, gam, bet, x, out);
}
